// BernoulliMonotonicAttention_46737834115653
// MI455X (gfx1250) — matrix operand order repaired and checked by reading, not compiled, not run on hardware
//
#include <hip/hip_runtime.h>
#include <hip/hip_bf16.h>
#include <hip/hip_fp16.h>

typedef __attribute__((ext_vector_type(16))) _Float16 v16h;
typedef __attribute__((ext_vector_type(8)))  float    v8f;

#define B_  32
#define L_  1024
#define DC_ 1024
#define H_  512
#define NEG_NUM (-10000.0f)

// branch-free fast tanh / sigmoid (exact saturation at +-inf of exp)
__device__ __forceinline__ float fast_tanh(float x) {
    float e = __expf(2.0f * x);
    return 1.0f - 2.0f * __builtin_amdgcn_rcpf(e + 1.0f);
}
__device__ __forceinline__ float fast_sigmoid(float x) {
    return __builtin_amdgcn_rcpf(1.0f + __expf(-x));
}

// ---------------------------------------------------------------------------
// Kernel 1: qproj[b][h] = b1[h] + sum_k query[b][k] * W1[DC+k][h]
// ---------------------------------------------------------------------------
__global__ void qproj_kernel(const float* __restrict__ query,
                             const float* __restrict__ W1,
                             const float* __restrict__ b1,
                             float* __restrict__ qproj) {
    const int h = threadIdx.x;          // 512 threads
    const int b = blockIdx.x;           // 32 blocks
    const float* W1q = W1 + (size_t)DC_ * H_;
    float acc = b1[h];
    #pragma unroll 4
    for (int k = 0; k < DC_; ++k)
        acc = fmaf(query[b * DC_ + k], W1q[(size_t)k * H_ + h], acc);
    qproj[b * H_ + h] = acc;
}

// ---------------------------------------------------------------------------
// Kernel 2: pack W1[:DC] (f32, [K=1024][N=512]) into f16 WMMA-B fragment
// order in workspace: w1p[nt(32)][kt(32)][lane(32)][16 f16]  (1 MB total).
// B frag per lane: n = lane&15, half = lane>>4; elems 0..7 = K[32*kt + half*8 + j], elems 8..15 = K[32*kt + 16 + half*8 + (j-8)]
// ---------------------------------------------------------------------------
__global__ void packW1_kernel(const float* __restrict__ W1,
                              _Float16* __restrict__ w1p) {
    int e = blockIdx.x * 256 + threadIdx.x;   // 2048 blocks -> 524288 elements
    int k = e >> 9;
    int n = e & 511;
    int nt = n >> 4, nn = n & 15;
    int kt = k >> 5, kk = k & 31;
    int hb = (kk >> 3) & 1, jb = (kk & 7) | ((kk >> 4) << 3);
    w1p[(((nt * 32 + kt) * 32) + (nn + 16 * hb)) * 16 + jb] =
        (_Float16)W1[(size_t)k * H_ + n];
}

// ---------------------------------------------------------------------------
// Kernel 3: main fused GEMM: p = sigmoid(mask( tanh(ctx@W1a + qproj)·w2 + b2 ) + noise)
// WG = 128 threads (4 waves). Each WG: 64 rows of M=B*L, full N=512, K=1024.
// LDS (dynamic, 192KB):
//   sA:  A fragments [kt(32)][mtile(4)][lane(32)][16 f16]          = 128KB
//   sB0/sB1: double-buffered B slabs [kt(32)][lane(32)][16 f16]    = 2x32KB
// B slabs are DMA'd pre-packed from workspace with async global->LDS copies
// (ASYNCcnt), prefetching slab nt+1 during the WMMAs of slab nt.
// K-loop: fully unrolled, with sched_group_barrier directives imposing a
// 1-stage software pipeline (ds_loads of kt+1 overlap the WMMA of kt).
// ---------------------------------------------------------------------------
#define SA_HALFS (32 * 4 * 32 * 16)   // 65536 f16 = 128KB
#define SB_HALFS (32 * 32 * 16)       // 16384 f16 = 32KB
#define SLAB_BYTES (SB_HALFS * 2)     // 32768 bytes
#define A_KT_STRIDE (4 * 32 * 16)     // f16 elems between kt planes of sA
#define B_KT_STRIDE (32 * 16)         // f16 elems between kt planes of sB
#define SCORE_LDS_BYTES (SA_HALFS * 2 + 2 * SLAB_BYTES)   // 196608

__device__ __forceinline__ void async_copy_slab(uint32_t ldsBase,
                                                const _Float16* gsrc,
                                                int tid) {
    uint64_t gbase = (uint64_t)(uintptr_t)gsrc;
    #pragma unroll
    for (int i = 0; i < 16; ++i) {
        uint32_t off = (uint32_t)(tid + i * 128) * 16u;   // 2048 x 16B = 32KB
        uint32_t lds = ldsBase + off;
        asm volatile("global_load_async_to_lds_b128 %0, %1, %2"
                     :: "v"(lds), "v"(off), "s"(gbase) : "memory");
    }
}

__global__ void score_kernel(const float* __restrict__ ctx,
                             const _Float16* __restrict__ w1p,
                             const float* __restrict__ w2,
                             const float* __restrict__ b2,
                             const int*   __restrict__ mask,
                             const float* __restrict__ noise,
                             const float* __restrict__ qproj,
                             float* __restrict__ p_out) {
    extern __shared__ __align__(16) char smem[];
    _Float16* sA = (_Float16*)smem;
    const uint32_t sBlds0 = (uint32_t)(uintptr_t)(sA + SA_HALFS);

    const int tid  = threadIdx.x;        // 0..127
    const int lane = tid & 31;
    const int mt   = tid >> 5;           // wave id -> its m-tile (0..3)
    const int row0 = blockIdx.x * 64;    // global M row of this WG
    const int b    = row0 >> 10;         // batch (64-row strip never crosses L)

    // ---- stage A: 64 x 1024 f32 -> f16, swizzled into WMMA A-fragment order
    // A frag per lane: m = lane&15, half = lane>>4;
    //   elems 0..7  = K[half*8 .. half*8+7], elems 8..15 = K[16+half*8 ..]
    for (int it = 0; it < 128; ++it) {
        int e  = tid + it * 128;         // 16384 float4 tiles: 64 rows x 256
        int r  = e >> 8;
        int k  = (e & 255) << 2;
        float4 v = *(const float4*)(ctx + (size_t)(row0 + r) * DC_ + k);
        int kt = k >> 5, kk = k & 31;
        int hf = (kk >> 3) & 1;
        int j  = (kk & 7) | ((kk >> 4) << 3);          // multiple of 4 here
        int am = r & 15, amt = r >> 4;
        _Float16* dst = sA + (((kt * 4 + amt) * 32) + (am + 16 * hf)) * 16 + j;
        dst[0] = (_Float16)v.x; dst[1] = (_Float16)v.y;
        dst[2] = (_Float16)v.z; dst[3] = (_Float16)v.w;
    }

    // ---- prefetch B slab for nt = 0
    async_copy_slab(sBlds0, w1p, tid);

    float sacc[8] = {0.f, 0.f, 0.f, 0.f, 0.f, 0.f, 0.f, 0.f};

    for (int nt = 0; nt < 32; ++nt) {
        // own wave's async copies of slab nt done; barrier makes all waves'
        // copies visible AND fences the buffer we are about to overwrite.
        asm volatile("s_wait_asynccnt 0x0" ::: "memory");
        __syncthreads();

        // kick off DMA of slab nt+1 into the other buffer (overlaps WMMAs)
        int ntn = (nt + 1) & 31;
        async_copy_slab(sBlds0 + (uint32_t)((nt + 1) & 1) * SLAB_BYTES,
                        w1p + (size_t)ntn * SB_HALFS, tid);

        // keep sB as plain arithmetic on the shared base so addrspace(3)
        // inference holds and these stay ds_load_b128 (not flat_load)
        const _Float16* aPtr = sA + (mt * 32 + lane) * 16;
        const _Float16* bPtr = sA + SA_HALFS + (nt & 1) * SB_HALFS + lane * 16;

        // ---- 32 WMMAs over K=1024 (fully unrolled)
        v8f acc = {};
        #pragma unroll
        for (int kt = 0; kt < 32; ++kt) {
            v16h afrag = *(const v16h*)(aPtr + kt * A_KT_STRIDE);
            v16h bfrag = *(const v16h*)(bPtr + kt * B_KT_STRIDE);
            acc = __builtin_amdgcn_wmma_f32_16x16x32_f16(
                false, afrag, false, bfrag, (short)0, acc, false, false);
        }
        // Impose 1-stage software pipeline: 2 fragment sets in flight so the
        // ds_loads of kt+1 overlap the WMMA of kt (s_wait_dscnt 4, not 0).
        // 0x200 = DS-read group, 0x008 = MFMA/WMMA group.
        __builtin_amdgcn_sched_group_barrier(0x200, 8, 0);     // kt0 + kt1 loads
        #pragma unroll
        for (int kt = 0; kt < 30; ++kt) {
            __builtin_amdgcn_sched_group_barrier(0x008, 1, 0); // WMMA kt
            __builtin_amdgcn_sched_group_barrier(0x200, 4, 0); // loads kt+2
        }
        __builtin_amdgcn_sched_group_barrier(0x008, 1, 0);     // WMMA 30
        __builtin_amdgcn_sched_group_barrier(0x008, 1, 0);     // WMMA 31

        // ---- epilogue: + qproj(+b1), tanh, * w2, reduce across 16 columns
        int ncol = nt * 16 + (lane & 15);
        float qp = qproj[b * H_ + ncol];
        float wv = w2[ncol];
        #pragma unroll
        for (int i = 0; i < 8; ++i) {
            float s = fast_tanh(acc[i] + qp) * wv;
            s += __shfl_xor(s, 1);
            s += __shfl_xor(s, 2);
            s += __shfl_xor(s, 4);
            s += __shfl_xor(s, 8);       // all 16 lanes of the half hold the sum
            sacc[i] += s;
        }
    }

    // ---- write p for this wave's 16 rows (lane 0 / lane 16 own 8 rows each)
    if ((lane & 15) == 0) {
        int hf = lane >> 4;
        float bias2 = b2[0];
        #pragma unroll
        for (int i = 0; i < 8; ++i) {
            int row = row0 + mt * 16 + i + 8 * hf;   // == b*L + l
            float sc = sacc[i] + bias2;
            if (mask[row] == 0) sc = NEG_NUM;
            sc += noise[row];
            p_out[row] = fast_sigmoid(sc);
        }
    }
}

// ---------------------------------------------------------------------------
// Kernel 4: att_t = p_t * prod_{i<t}(1 - p_i)   (wave-parallel prefix product)
// ---------------------------------------------------------------------------
__global__ void scan_kernel(const float* __restrict__ p,
                            float* __restrict__ att_out) {
    const int b = blockIdx.x;
    const int lane = threadIdx.x;        // 32 lanes, one wave
    float carry = 1.0f;
    for (int c = 0; c < 32; ++c) {
        int l = c * 32 + lane;
        float pl = p[b * L_ + l];
        float q = 1.0f - pl;
        float incl = q;
        #pragma unroll
        for (int off = 1; off < 32; off <<= 1) {
            float v = __shfl_up(incl, off);
            if (lane >= off) incl *= v;
        }
        float excl = __shfl_up(incl, 1);
        if (lane == 0) excl = 1.0f;
        att_out[b * L_ + l] = carry * excl * pl;
        carry *= __shfl(incl, 31);
    }
}

// ---------------------------------------------------------------------------
// Kernel 5: expected_ctx[b][c] = sum_l att[b][l] * ctx[b][l][c]  (BW-bound)
// ---------------------------------------------------------------------------
__global__ void ectx_kernel(const float* __restrict__ att,
                            const float* __restrict__ ctx,
                            float* __restrict__ out) {
    const int b = blockIdx.y;
    const int c = blockIdx.x * 256 + threadIdx.x;
    const float* cp = ctx + (size_t)b * L_ * DC_ + c;
    const float* ap = att + b * L_;
    float acc = 0.0f;
    #pragma unroll 8
    for (int l = 0; l < L_; ++l)
        acc = fmaf(ap[l], cp[(size_t)l * DC_], acc);
    out[b * DC_ + c] = acc;
}

// ---------------------------------------------------------------------------
extern "C" void kernel_launch(void* const* d_in, const int* in_sizes, int n_in,
                              void* d_out, int out_size, void* d_ws, size_t ws_size,
                              hipStream_t stream) {
    const float* ctx   = (const float*)d_in[0];
    const float* query = (const float*)d_in[1];
    const int*   mask  = (const int*)d_in[2];
    const float* noise = (const float*)d_in[3];
    const float* W1    = (const float*)d_in[4];
    const float* b1    = (const float*)d_in[5];
    const float* w2    = (const float*)d_in[6];
    const float* b2    = (const float*)d_in[7];

    float* out   = (float*)d_out;              // [0 .. 32K): expected_ctx, [32K .. 64K): att
    float* att   = out + B_ * L_;
    float*     qproj = (float*)d_ws;                       // 32*512 f32
    float*     p     = qproj + B_ * H_;                    // 32*1024 f32
    _Float16*  w1p   = (_Float16*)(p + B_ * L_);           // 1024*512 f16 (1MB)

    qproj_kernel<<<B_, H_, 0, stream>>>(query, W1, b1, qproj);
    packW1_kernel<<<(DC_ * H_) / 256, 256, 0, stream>>>(W1, w1p);
    score_kernel<<<(B_ * L_) / 64, 128, SCORE_LDS_BYTES, stream>>>(
        ctx, w1p, w2, b2, mask, noise, qproj, p);
    scan_kernel<<<B_, 32, 0, stream>>>(p, att);
    ectx_kernel<<<dim3(DC_ / 256, B_), 256, 0, stream>>>(att, ctx, out);
}
